// PairContrastiveLoss_69174743269973
// MI455X (gfx1250) — matrix operand order repaired and checked by reading, not compiled, not run on hardware
//
#include <hip/hip_runtime.h>

typedef __attribute__((ext_vector_type(16))) _Float16 v16h;
typedef __attribute__((ext_vector_type(8)))  float    v8f;

#define HIDDEN   768
#define NP       196
#define NPAD     208     // 13 * 16
#define KCHUNK   96      // K chunk resident in LDS (3 WMMA k-steps)
#define KV8      (KCHUNK / 8)   // 12 uint4 per column per chunk
#define NCHUNKS  8       // 768 / 96
#define HW       50176   // 224*224
#define NB       4
#define EPSN     1e-8f

// ws layout (floats):
//  wsF[0..5]  = g00,g01,g02,g11,g12,g22   (Gram of pixel_w, unique entries)
//  wsF[6..8]  = Wt*b
//  wsF[9]     = b*b
//  wsF[15]    = global accumulator
//  wsF[16..]  = float4 wp4[768]  (w0,w1,w2,bias per hidden dim)   -> 12288 B
//  byte off 12544: _Float16 qn[4][208][768]  (normalized patches) -> ~1.22 MB
#define WS_QN_FLOAT_OFF 3136   // 12544 / 4

// ---------------------------------------------------------------------------
// Kernel 0: Gram matrix / packed weight table / accumulator init
// ---------------------------------------------------------------------------
__global__ __launch_bounds__(256) void prep_kernel(
    const float* __restrict__ pixel_w,
    const float* __restrict__ pixel_b,
    float* __restrict__ wsF) {
  __shared__ float red[256];
  const int tid = threadIdx.x;
  float s[10];
  #pragma unroll
  for (int i = 0; i < 10; ++i) s[i] = 0.f;

  float4* wp4 = (float4*)(wsF + 16);
  for (int d = tid; d < HIDDEN; d += 256) {
    float w0 = pixel_w[d * 3 + 0];
    float w1 = pixel_w[d * 3 + 1];
    float w2 = pixel_w[d * 3 + 2];
    float bb = pixel_b[d];
    s[0] += w0 * w0; s[1] += w0 * w1; s[2] += w0 * w2;
    s[3] += w1 * w1; s[4] += w1 * w2; s[5] += w2 * w2;
    s[6] += w0 * bb; s[7] += w1 * bb; s[8] += w2 * bb;
    s[9] += bb * bb;
    wp4[d] = make_float4(w0, w1, w2, bb);
  }
  for (int i = 0; i < 10; ++i) {
    red[tid] = s[i];
    __syncthreads();
    for (int off = 128; off > 0; off >>= 1) {
      if (tid < off) red[tid] += red[tid + off];
      __syncthreads();
    }
    if (tid == 0) wsF[i] = red[0];
    __syncthreads();
  }
  if (tid == 0) wsF[15] = 0.f;
}

// ---------------------------------------------------------------------------
// Kernel 1: patch embedding (16x16 stride-16 conv), row-normalize, f16 store
// One block per (batch, patch): 784 blocks x 256 threads
// ---------------------------------------------------------------------------
__global__ __launch_bounds__(256) void patch_kernel(
    const float* __restrict__ x,
    const float* __restrict__ patch_w,
    const float* __restrict__ patch_b,
    float* __restrict__ wsF) {
  __shared__ float xp[768];
  __shared__ float red[256];
  const int tid = threadIdx.x;
  const int blk = blockIdx.x;          // b*196 + p
  const int b = blk / NP, p = blk % NP;
  const int ph = p / 14, pw = p % 14;

  // stage the 3x16x16 input patch in the same layout as patch_w rows
  for (int i = tid; i < 768; i += 256) {
    const int c = i >> 8, r = (i >> 4) & 15, q = i & 15;
    xp[i] = x[(((b * 3 + c) * 224) + ph * 16 + r) * 224 + pw * 16 + q];
  }
  __syncthreads();

  float v[3];
  float ss = 0.f;
  #pragma unroll
  for (int j = 0; j < 3; ++j) {
    const int d = tid + j * 256;
    const float* wrow = patch_w + (size_t)d * 768;
    float acc = patch_b[d];
    #pragma unroll 8
    for (int k = 0; k < 768; ++k) acc = fmaf(xp[k], wrow[k], acc);
    v[j] = acc;
    ss += acc * acc;
  }

  red[tid] = ss;
  __syncthreads();
  for (int off = 128; off > 0; off >>= 1) {
    if (tid < off) red[tid] += red[tid + off];
    __syncthreads();
  }
  const float invn = 1.0f / fmaxf(sqrtf(red[0]), EPSN);

  _Float16* qn = (_Float16*)(wsF + WS_QN_FLOAT_OFF);
  const size_t row = (size_t)(b * NPAD + p) * HIDDEN;
  #pragma unroll
  for (int j = 0; j < 3; ++j)
    qn[row + tid + j * 256] = (_Float16)(v[j] * invn);
}

// ---------------------------------------------------------------------------
// Kernel 2: fused pixel-embed + normalize + WMMA similarity + masked exp-sum
// grid (784, 4), block 256 (8 waves).
// wave = (mtile, nhalf): 16 pixel rows x 7 patch tiles (tile 6 overlaps,
// counted once in the reduction). 56 accumulator VGPRs per lane -> no spills.
// qn K-chunks staged into LDS with global_load_async_to_lds_b128 (ASYNCcnt).
// ---------------------------------------------------------------------------
__global__ __launch_bounds__(256) void sim_kernel(
    const float* __restrict__ x,
    float* __restrict__ wsF) {
  __shared__ float4 swp[HIDDEN];                                       // 12288 B
  __shared__ __attribute__((aligned(32))) _Float16 sqn[NPAD * KCHUNK]; // 39936 B
  __shared__ float sred[8];

  const int tid   = threadIdx.x;
  const int lane  = tid & 31;
  const int wave  = tid >> 5;
  const int mtile = wave >> 1;     // 0..3 : which 16-row group
  const int nhalf = wave & 1;      // 0 -> patch tiles 0..6, 1 -> tiles 6..12
  const int b     = blockIdx.y;
  const int tile  = blockIdx.x;    // 64 pixel rows per block

  // uniform prep constants
  const float g00 = wsF[0], g01 = wsF[1], g02 = wsF[2];
  const float g11 = wsF[3], g12 = wsF[4], g22 = wsF[5];
  const float wb0 = wsF[6], wb1 = wsF[7], wb2 = wsF[8];
  const float bbq = wsF[9];

  // stage packed (w0,w1,w2,bias) table
  const float4* wp4g = (const float4*)(wsF + 16);
  for (int i = tid; i < HIDDEN; i += 256) swp[i] = wp4g[i];

  // zero the 12 padded columns once; async loads never touch them
  {
    const uint4 z = make_uint4(0u, 0u, 0u, 0u);
    uint4* dstp = (uint4*)sqn;
    for (int i = tid; i < (NPAD - NP) * KV8; i += 256) {
      const int p  = NP + i / KV8;
      const int kv = i % KV8;
      dstp[p * KV8 + kv] = z;
    }
  }

  // this lane's pixel row (A fragment: both lane halves carry M = lane%16)
  const int mrow    = lane & 15;
  const int rowBase = tile * 64 + mtile * 16;
  const int n       = rowBase + mrow;          // pixel index in batch
  const float* xb = x + (size_t)b * 3 * HW;
  const float x0 = xb[n];
  const float x1 = xb[HW + n];
  const float x2 = xb[2 * HW + n];
  const float qq = g00 * x0 * x0 + g11 * x1 * x1 + g22 * x2 * x2 +
                   2.f * (g01 * x0 * x1 + g02 * x0 * x2 + g12 * x1 * x2) +
                   2.f * (wb0 * x0 + wb1 * x1 + wb2 * x2) + bbq;
  const float invn = 1.0f / fmaxf(sqrtf(fmaxf(qq, 0.f)), EPSN);

  const int laneHalf = (lane < 16) ? 0 : 16;   // A-matrix K sub-offset
  const int khalfB   = (lane < 16) ? 0 : 16;   // B-matrix K sub-offset

  v8f zero = {};
  v8f acc[7];
  #pragma unroll
  for (int t = 0; t < 7; ++t) acc[t] = zero;

  const _Float16* qn =
      (const _Float16*)(wsF + WS_QN_FLOAT_OFF) + (size_t)b * NPAD * HIDDEN;
  const uint4* qnv = (const uint4*)qn;               // 8 halves per uint4
  const unsigned ldsBase = (unsigned)(uintptr_t)sqn; // LDS offset in low 32b

  for (int ch = 0; ch < NCHUNKS; ++ch) {
    __syncthreads();  // previous chunk fully consumed (and pad zeros visible)

    // async DMA the qn K-chunk straight into LDS (196 cols x 96 halves)
    for (int idx = tid; idx < NP * KV8; idx += 256) {
      const int p  = idx / KV8;
      const int kv = idx % KV8;
      const uint4* gp = qnv + (size_t)p * (HIDDEN / 8) + ch * KV8 + kv;
      const unsigned la = ldsBase + (unsigned)(p * KV8 + kv) * 16u;
      asm volatile("global_load_async_to_lds_b128 %0, %1, off"
                   :: "v"(la), "v"(gp) : "memory");
    }
    asm volatile("s_wait_asynccnt 0x0" ::: "memory");
    __syncthreads();

    #pragma unroll
    for (int kk = 0; kk < KCHUNK; kk += 32) {
      // build A fragment in registers: pn[m, k] = (W[k,:]·x + b[k]) * invn
      v16h afrag;
      const int dbase = ch * KCHUNK + kk;
      #pragma unroll
      for (int j = 0; j < 8; ++j) {
        const int d0 = dbase + laneHalf + j;          // K = laneHalf + j
        const float4 w0v = swp[d0];
        const float v0 =
            fmaf(x2, w0v.z, fmaf(x1, w0v.y, fmaf(x0, w0v.x, w0v.w))) * invn;
        afrag[j] = (_Float16)v0;
        const int d1 = dbase + 8 + laneHalf + j;      // K = 8 + laneHalf + j
        const float4 w1v = swp[d1];
        const float v1 =
            fmaf(x2, w1v.z, fmaf(x1, w1v.y, fmaf(x0, w1v.x, w1v.w))) * invn;
        afrag[8 + j] = (_Float16)v1;
      }

      #pragma unroll
      for (int tt = 0; tt < 7; ++tt) {
        const int ptile = nhalf * 6 + tt;             // 0..6 or 6..12
        const int pcol  = ptile * 16 + (lane & 15);   // B: N = lane%16
        const v16h* bp =
            (const v16h*)(sqn + pcol * KCHUNK + kk + khalfB);
        v16h bfrag = *bp;
        acc[tt] = __builtin_amdgcn_wmma_f32_16x16x32_f16(
            false, afrag, false, bfrag, (short)0, acc[tt], false, false);
      }
    }
  }

  // masked exp-sum: C/D layout -> M = r + (lane<16?0:8), N = t*16 + lane%16
  float sum = 0.f;
  const int rowLane = (lane < 16) ? 0 : 8;
  #pragma unroll
  for (int tt = 0; tt < 7; ++tt) {
    if (nhalf && tt == 0) continue;   // overlapped tile 6 counted by nhalf==0
    const int ptile = nhalf * 6 + tt;
    const int pcol  = ptile * 16 + (lane & 15);
    #pragma unroll
    for (int r = 0; r < 8; ++r) {
      const int nn = rowBase + r + rowLane;
      if (pcol < NP && pcol != (nn % NP))
        sum += __expf(2.0f * acc[tt][r]);
    }
  }

  // wave32 shuffle reduce, then cross-wave via LDS, one atomic per block
  #pragma unroll
  for (int off = 16; off > 0; off >>= 1)
    sum += __shfl_xor(sum, off, 32);
  if (lane == 0) sred[wave] = sum;
  __syncthreads();
  if (tid == 0) {
    float tot = 0.f;
    #pragma unroll
    for (int i = 0; i < 8; ++i) tot += sred[i];
    atomicAdd(&wsF[15], tot);
  }
}

// ---------------------------------------------------------------------------
// Kernel 3: final log
// ---------------------------------------------------------------------------
__global__ void finalize_kernel(const float* __restrict__ wsF,
                                float* __restrict__ out) {
  if (threadIdx.x == 0) out[0] = logf(wsF[15]);
}

// ---------------------------------------------------------------------------
extern "C" void kernel_launch(void* const* d_in, const int* in_sizes, int n_in,
                              void* d_out, int out_size, void* d_ws,
                              size_t ws_size, hipStream_t stream) {
  const float* x       = (const float*)d_in[0];
  const float* pixel_w = (const float*)d_in[1];
  const float* pixel_b = (const float*)d_in[2];
  const float* patch_w = (const float*)d_in[3];
  const float* patch_b = (const float*)d_in[4];
  float* wsF = (float*)d_ws;
  float* out = (float*)d_out;

  prep_kernel<<<1, 256, 0, stream>>>(pixel_w, pixel_b, wsF);
  patch_kernel<<<NB * NP, 256, 0, stream>>>(x, patch_w, patch_b, wsF);
  dim3 grid(HW / 64, NB);
  sim_kernel<<<grid, 256, 0, stream>>>(x, wsF);
  finalize_kernel<<<1, 64, 0, stream>>>(wsF, out);
}
